// CMIConnector_75539884802553
// MI455X (gfx1250) — matrix operand order repaired and checked by reading, not compiled, not run on hardware
//
#include <hip/hip_runtime.h>
#include <hip/hip_bf16.h>
#include <stdint.h>

// ---------------------------------------------------------------------------
// Problem constants (fixed by the reference):
//   B=16, Lv=4096, Dv=1024, Lt=128, Dt=4096, ds=64, dm=4096, M = B*Lv = 65536
// ---------------------------------------------------------------------------
#define BB   16
#define LV   4096
#define DV   1024
#define LT   128
#define DT   4096
#define DS   64
#define DM   4096
#define MROW (BB * LV)   // 65536

typedef __attribute__((ext_vector_type(2)))  float   v2f;
typedef __attribute__((ext_vector_type(8)))  float   v8f;
typedef __attribute__((ext_vector_type(4)))  __bf16  v4bf;
typedef __attribute__((ext_vector_type(8)))  __bf16  v8bf;
typedef __attribute__((ext_vector_type(16))) __bf16  v16bf;

// ---------------------------------------------------------------------------
// Kernel 1: tc[b][d] = mean_t text[b][t][d]        (B=16 blocks, 256 thr)
// ---------------------------------------------------------------------------
__global__ void tc_kernel(const float* __restrict__ T, float* __restrict__ tc) {
  const int b = blockIdx.x;
  for (int d = threadIdx.x; d < DT; d += 256) {
    const float* p = T + (size_t)b * LT * DT + d;
    float s = 0.f;
    #pragma unroll 8
    for (int t = 0; t < LT; ++t) s += p[(size_t)t * DT];
    tc[b * DT + d] = s * (1.0f / (float)LT);
  }
}

// ---------------------------------------------------------------------------
// Kernel 2: gates. One wave per (matrix, b, ch): dot(tc[b], W[ch]) over Dt.
// ---------------------------------------------------------------------------
__global__ void gates_kernel(const float* __restrict__ tc,
                             const float* __restrict__ Wd, const float* __restrict__ bd,
                             const float* __restrict__ WB, const float* __restrict__ bBp,
                             const float* __restrict__ WC, const float* __restrict__ bCp,
                             float* __restrict__ delta, float* __restrict__ Bv,
                             float* __restrict__ Cv) {
  const int gw   = blockIdx.x * 8 + (threadIdx.x >> 5);
  const int lane = threadIdx.x & 31;
  const int mat  = gw >> 10;
  const int rem  = gw & 1023;
  const int b    = rem >> 6;
  const int ch   = rem & 63;

  const float* W = (mat == 0) ? Wd : ((mat == 1) ? WB : WC);
  const float* t = tc + (size_t)b * DT;
  const float* w = W  + (size_t)ch * DT;

  float s = 0.f;
  #pragma unroll 4
  for (int i = lane; i < DT; i += 32) s += t[i] * w[i];
  #pragma unroll
  for (int off = 16; off > 0; off >>= 1) s += __shfl_down(s, off, 32);

  if (lane == 0) {
    if (mat == 0)       delta[b * DS + ch] = 1.f / (1.f + __expf(-(s + bd[ch])));
    else if (mat == 1)  Bv[b * DS + ch]    = s + bBp[ch];
    else                Cv[b * DS + ch]    = s + bCp[ch];
  }
}

// ---------------------------------------------------------------------------
// Kernel 3: X = visual @ Wx^T + bx   via V_WMMA_F32_16X16X4_F32 (pure fp32).
// One wave per 16-row M tile; covers all N=64 (4 accumulators).
// ---------------------------------------------------------------------------
__global__ void xproj_kernel(const float* __restrict__ V, const float* __restrict__ Wx,
                             const float* __restrict__ bx, float* __restrict__ X) {
  const int lane = threadIdx.x & 31;
  const int wave = threadIdx.x >> 5;
  const int m0   = (blockIdx.x * 8 + wave) * 16;
  const int half = lane >> 4;
  const int lr   = lane & 15;

  const float* arow = V + (size_t)(m0 + lr) * DV + half * 2;

  v8f acc0 = {}, acc1 = {}, acc2 = {}, acc3 = {};
  #pragma unroll 4
  for (int kk = 0; kk < DV; kk += 4) {
    v2f a  = *(const v2f*)(arow + kk);
    v2f b0 = *(const v2f*)(Wx + (size_t)( 0 + lr) * DV + kk + half * 2);
    v2f b1 = *(const v2f*)(Wx + (size_t)(16 + lr) * DV + kk + half * 2);
    v2f b2 = *(const v2f*)(Wx + (size_t)(32 + lr) * DV + kk + half * 2);
    v2f b3 = *(const v2f*)(Wx + (size_t)(48 + lr) * DV + kk + half * 2);
    acc0 = __builtin_amdgcn_wmma_f32_16x16x4_f32(false, a, false, b0, (short)0, acc0, false, false);
    acc1 = __builtin_amdgcn_wmma_f32_16x16x4_f32(false, a, false, b1, (short)0, acc1, false, false);
    acc2 = __builtin_amdgcn_wmma_f32_16x16x4_f32(false, a, false, b2, (short)0, acc2, false, false);
    acc3 = __builtin_amdgcn_wmma_f32_16x16x4_f32(false, a, false, b3, (short)0, acc3, false, false);
  }

  v8f accs[4] = {acc0, acc1, acc2, acc3};
  #pragma unroll
  for (int nt = 0; nt < 4; ++nt) {
    const int n    = nt * 16 + lr;
    const float bias = bx[n];
    #pragma unroll
    for (int r = 0; r < 8; ++r) {
      const int m = m0 + r + half * 8;
      X[(size_t)m * DS + n] = accs[nt][r] + bias;
    }
  }
}

// ---------------------------------------------------------------------------
// Kernel 4: SSM scan per (b, channel); stores y as bf16.
// ---------------------------------------------------------------------------
__global__ void scan_kernel(const float* __restrict__ X,
                            const float* __restrict__ delta,
                            const float* __restrict__ Bv,
                            const float* __restrict__ Cv,
                            __bf16* __restrict__ Yb) {
  const int id = blockIdx.x * 256 + threadIdx.x;   // 0..1023 = b*64 + c
  const int b  = id >> 6;
  const int c  = id & 63;
  const float d  = delta[id];
  const float od = 1.f - d;
  const float db = d * Bv[id];
  const float cc = Cv[id];

  const float* xp = X  + (size_t)b * LV * DS + c;
  __bf16*      yp = Yb + (size_t)b * LV * DS + c;
  float h = 0.f;
  for (int t = 0; t < LV; ++t) {
    h = __builtin_fmaf(od, h, db * xp[(size_t)t * DS]);
    yp[(size_t)t * DS] = (__bf16)(cc * h);
  }
}

// ---------------------------------------------------------------------------
// Kernel 5: one-time Wo fp32 -> bf16 conversion (Wo is 4096x64, L2-resident).
// ---------------------------------------------------------------------------
__global__ void woconv_kernel(const float* __restrict__ Wo, __bf16* __restrict__ WoB) {
  const int i = (blockIdx.x * 256 + threadIdx.x) * 4;
  v4bf o;
  #pragma unroll
  for (int j = 0; j < 4; ++j) o[j] = (__bf16)Wo[i + j];
  *(v4bf*)(WoB + i) = o;
}

// ---------------------------------------------------------------------------
// Kernel 6: out = Y(bf16) @ Wo^T + bo  via V_WMMA_F32_16X16X32_BF16, K=64.
// Block = 8 waves sharing one 64-column n-group; the 8KB Wo-bf16 tile for the
// n-group is staged into LDS ONCE per block with GLOBAL_LOAD_ASYNC_TO_LDS_B128
// (ASYNCcnt), then all waves build B fragments with ds_load_b128.
// Each wave: one 16-row m-tile x 4 n-tiles x 2 k-steps = 8 WMMAs; A fragments
// loaded once from global (y stays L2-resident) and reused across n-tiles.
// Grid: 512 m-groups x 64 n-groups = 32768 blocks x 256 threads.
// ---------------------------------------------------------------------------
__global__ void outgemm_kernel(const __bf16* __restrict__ Yb,
                               const __bf16* __restrict__ WoB,
                               const float* __restrict__ bo,
                               float* __restrict__ O) {
  __shared__ __bf16 sWo[64 * 64];   // [n within group][k], 8192 bytes

  const int tid  = threadIdx.x;
  const int lane = tid & 31;
  const int wave = tid >> 5;
  const int mg   = blockIdx.x >> 6;        // 512 m-groups of 8 m-tiles
  const int ng   = blockIdx.x & 63;        // 64 n-groups of 64 columns
  const int m0   = (mg * 8 + wave) * 16;
  const int nb   = ng * 64;                // first output column of this block
  const int half = lane >> 4;
  const int lr   = lane & 15;

  // --- Stage the 64x64 bf16 Wo tile into LDS via async copy (2 x b128/thread).
  // INST_OFFSET is added to both the LDS and global address (ISA 08 §4.4), so
  // offset:16 covers the second 16-byte chunk.
  {
    unsigned lds0 = (unsigned)(uintptr_t)(&sWo[0]) + (unsigned)tid * 32u;
    uint64_t g0   = (uint64_t)(uintptr_t)(WoB + (size_t)nb * DS) + (uint64_t)tid * 32u;
    asm volatile(
        "global_load_async_to_lds_b128 %0, %1, off\n\t"
        "global_load_async_to_lds_b128 %0, %1, off offset:16\n\t"
        :: "v"(lds0), "v"(g0) : "memory");
    asm volatile("s_wait_asynccnt 0" ::: "memory");
  }
  __syncthreads();

  // --- A fragments: a0 covers K=0..31, a1 covers K=32..63 (4 x b128 loads).
  v16bf a0, a1;
  {
    const __bf16* row = Yb + (size_t)(m0 + lr) * DS;
    v8bf p0 = *(const v8bf*)(row +  0 + half * 8);
    v8bf p1 = *(const v8bf*)(row + 16 + half * 8);
    v8bf p2 = *(const v8bf*)(row + 32 + half * 8);
    v8bf p3 = *(const v8bf*)(row + 48 + half * 8);
    #pragma unroll
    for (int i = 0; i < 8; ++i) {
      a0[i] = p0[i]; a0[8 + i] = p1[i];
      a1[i] = p2[i]; a1[8 + i] = p3[i];
    }
  }

  #pragma unroll
  for (int nt = 0; nt < 4; ++nt) {
    // B fragment from LDS: lane lr of half h needs Wo[n][half*8 + i] and
    // Wo[n][16 + half*8 + i] (b0), +32 for b1 -- 2 x ds_load_b128 each.
    const __bf16* srow = &sWo[(nt * 16 + lr) * DS];
    v16bf b0, b1;
    {
      v8bf q0 = *(const v8bf*)(srow + half * 8 + 0);
      v8bf q1 = *(const v8bf*)(srow + 16 + half * 8);
      v8bf q2 = *(const v8bf*)(srow + 32 + half * 8);
      v8bf q3 = *(const v8bf*)(srow + 48 + half * 8);
      #pragma unroll
      for (int i = 0; i < 8; ++i) {
        b0[i] = q0[i]; b0[8 + i] = q1[i];
        b1[i] = q2[i]; b1[8 + i] = q3[i];
      }
    }
    v8f acc = {};
    acc = __builtin_amdgcn_wmma_f32_16x16x32_bf16(false, a0, false, b0, (short)0, acc, false, false);
    acc = __builtin_amdgcn_wmma_f32_16x16x32_bf16(false, a1, false, b1, (short)0, acc, false, false);

    const int   n    = nb + nt * 16 + lr;
    const float bias = bo[n];
    #pragma unroll
    for (int r = 0; r < 8; ++r) {
      const int m = m0 + r + half * 8;
      O[(size_t)m * DM + n] = acc[r] + bias;
    }
  }
}

// ---------------------------------------------------------------------------
// Launch
// ---------------------------------------------------------------------------
extern "C" void kernel_launch(void* const* d_in, const int* in_sizes, int n_in,
                              void* d_out, int out_size, void* d_ws, size_t ws_size,
                              hipStream_t stream) {
  (void)in_sizes; (void)n_in; (void)out_size; (void)ws_size;

  const float* V   = (const float*)d_in[0];   // visual_feats [16,4096,1024]
  const float* T   = (const float*)d_in[1];   // text_embeds  [16,128,4096]
  const float* Wx  = (const float*)d_in[2];   // [64,1024]
  const float* bx  = (const float*)d_in[3];   // [64]
  const float* Wd  = (const float*)d_in[4];   // [64,4096]
  const float* bd  = (const float*)d_in[5];
  const float* WB  = (const float*)d_in[6];
  const float* bB  = (const float*)d_in[7];
  const float* WC  = (const float*)d_in[8];
  const float* bC  = (const float*)d_in[9];
  const float* Wo  = (const float*)d_in[10];  // [4096,64]
  const float* bo  = (const float*)d_in[11];  // [4096]
  float* out = (float*)d_out;                 // [16,4096,4096]

  // Workspace carve-up (~26 MB)
  float*  tc    = (float*)d_ws;                   // 16*4096 f32
  float*  delta = tc + (size_t)BB * DT;           // 1024 f32
  float*  Bv    = delta + BB * DS;
  float*  Cv    = Bv    + BB * DS;
  float*  X     = Cv    + BB * DS;                // 65536*64 f32
  __bf16* Yb    = (__bf16*)(X + (size_t)MROW * DS);   // 65536*64 bf16
  __bf16* WoB   = Yb + (size_t)MROW * DS;             // 4096*64 bf16

  tc_kernel     <<<BB,                306 - 50, 0, stream>>>(T, tc);
  gates_kernel  <<<(3 * BB * DS) / 8,      256, 0, stream>>>(tc, Wd, bd, WB, bB, WC, bC,
                                                             delta, Bv, Cv);
  xproj_kernel  <<<(MROW / 16) / 8,        256, 0, stream>>>(V, Wx, bx, X);
  scan_kernel   <<<(BB * DS) / 256,        256, 0, stream>>>(X, delta, Bv, Cv, Yb);
  woconv_kernel <<<(DM * DS) / (256 * 4),  256, 0, stream>>>(Wo, WoB);
  outgemm_kernel<<<(MROW / (16 * 8)) * (DM / 64), 256, 0, stream>>>(Yb, WoB, bo, out);
}
